// RLCA_66554813219284
// MI455X (gfx1250) — matrix operand order repaired and checked by reading, not compiled, not run on hardware
//
#include <hip/hip_runtime.h>
#include <hip/hip_bf16.h>

typedef __attribute__((ext_vector_type(16))) _Float16 v16h;
typedef __attribute__((ext_vector_type(8)))  _Float16 v8h;
typedef __attribute__((ext_vector_type(8)))  float    v8f;
typedef __attribute__((ext_vector_type(4)))  unsigned int v4u;
typedef __attribute__((ext_vector_type(8)))  int      v8i;
typedef __attribute__((ext_vector_type(4)))  int      v4i;

#define BSZ    4
#define NQ     512
#define MK     4096
#define DMODEL 512
#define NHEAD  8
#define DHEAD  64
#define NR     31            // 2*RPR_LEN - 1
#define SIGMA_INV (1.0f/1.5f)
#define EXP_M2SIG 0.26359713f   // exp(-2/1.5)
#define NEGINF (-1.0e30f)

// -------- WMMA helper: D = A(16x32 f16) * B(32x16 f16) + C(f32) ------------
__device__ __forceinline__ v8f wmma16(v16h a, v16h b, v8f c) {
  return __builtin_amdgcn_wmma_f32_16x16x32_f16(
      /*neg_a=*/false, a, /*neg_b=*/false, b,
      /*c_mod=*/(short)0, c, /*reuse_a=*/false, /*reuse_b=*/false);
}

// A-fragment load in contiguous k order: lane holds row (lane&15), K = kbase + {0..15}
__device__ __forceinline__ v16h load_afrag(const _Float16* p) {
  const v8h* q = (const v8h*)p;
  v8h lo = q[0];
  v8h hi = q[1];      // +8 elements
  v16h r;
#pragma unroll
  for (int j = 0; j < 8; ++j) { r[j] = lo[j]; r[8 + j] = hi[j]; }
  return r;
}

// ---------------------------- f32 -> f16 convert ---------------------------
__global__ void cvt_f16_kernel(const float* __restrict__ in,
                               _Float16* __restrict__ out, int n) {
  int i = (blockIdx.x * blockDim.x + threadIdx.x) * 4;
  if (i >= n) return;
  float4 v = *(const float4*)(in + i);
  out[i + 0] = (_Float16)v.x;
  out[i + 1] = (_Float16)v.y;
  out[i + 2] = (_Float16)v.z;
  out[i + 3] = (_Float16)v.w;
}

// ------------------- W[k,c] (f32) -> Wt[c,k] (f16) transpose ---------------
__global__ void wtrans_kernel(const float* __restrict__ W,
                              _Float16* __restrict__ Wt) {
  __shared__ float tile[32][33];
  int k0 = blockIdx.x * 32, c0 = blockIdx.y * 32;
  int tx = threadIdx.x & 31, ty = threadIdx.x >> 5;   // 256 threads
  for (int r = ty; r < 32; r += 8)
    tile[r][tx] = W[(size_t)(k0 + r) * DMODEL + c0 + tx];
  __syncthreads();
  for (int r = ty; r < 32; r += 8)
    Wt[(size_t)(c0 + r) * DMODEL + k0 + tx] = (_Float16)tile[tx][r];
}

// ------------- GEMM: Out[row,col] = sum_k X[row,k]*Wt[col,k] + bias[col] ---
// 128 threads = 4 waves; wave w owns col tile blockIdx.y*64 + w*16.
template <bool OUTF32>
__global__ void __launch_bounds__(128)
gemm_proj(const _Float16* __restrict__ X, const _Float16* __restrict__ Wt,
          const float* __restrict__ bias, void* __restrict__ out, int kdim,
          int cols) {
  int lane = threadIdx.x & 31;
  int wave = threadIdx.x >> 5;
  int half = lane >> 4;
  int l16  = lane & 15;
  int kbaseA = half * 16;
  int kbaseB = half * 16;
  size_t row = (size_t)blockIdx.x * 16 + l16;
  int col = blockIdx.y * 64 + wave * 16 + l16;

  const _Float16* xp = X + row * (size_t)kdim + kbaseA;
  const _Float16* wp = Wt + (size_t)col * kdim + kbaseB;

  v8f c = {0.f, 0.f, 0.f, 0.f, 0.f, 0.f, 0.f, 0.f};
  for (int k0 = 0; k0 < kdim; k0 += 32) {
    v16h a = load_afrag(xp + k0);
    v16h b = *(const v16h*)(wp + k0);
    c = wmma16(a, b, c);
  }
  float bb = bias[col];
  size_t rbase = (size_t)blockIdx.x * 16 + half * 8;
#pragma unroll
  for (int v = 0; v < 8; ++v) {
    float val = c[v] + bb;
    size_t r = rbase + v;
    if (OUTF32) ((float*)out)[r * cols + col] = val;
    else        ((_Float16*)out)[r * cols + col] = (_Float16)val;
  }
}

// -------------------- QEr[b,h,n,r] = Q[b,n,h,:] . Er[r,:] ------------------
__global__ void qer_kernel(const _Float16* __restrict__ Qf,
                           const float* __restrict__ Er,
                           float* __restrict__ QErg) {
  int idx = blockIdx.x;                 // b*NHEAD*NQ blocks
  int n = idx % NQ;
  int h = (idx / NQ) % NHEAD;
  int b = idx / (NQ * NHEAD);
  int r = threadIdx.x;
  if (r >= NR) return;
  const _Float16* q = Qf + ((size_t)(b * NQ + n)) * DMODEL + h * DHEAD;
  const float* e = Er + (size_t)r * DHEAD;
  float s = 0.f;
#pragma unroll 8
  for (int d = 0; d < DHEAD; ++d) s += (float)q[d] * e[d];
  QErg[(((size_t)(b * NHEAD + h)) * NQ + n) * NR + r] = s;
}

// --------------------- fused flash attention + rel bias --------------------
// block = 128 threads (4 waves): 16 query rows of one (b,h); keys in chunks
// of 64.  K chunk staged into LDS by the Tensor Data Mover; V chunk staged
// transposed by the waves.  Wave w owns key-cols / out-cols w*16..w*16+15.
__global__ void __launch_bounds__(128)
attn_kernel(const _Float16* __restrict__ Qf, const _Float16* __restrict__ Kf,
            const _Float16* __restrict__ Vf, const float* __restrict__ QErg,
            _Float16* __restrict__ Og) {
  __shared__ float    s_s[16][64];
  __shared__ _Float16 s_p[16 * 64];
  __shared__ float    s_qer[16][NR];
  __shared__ float    s_corr[16];
  __shared__ _Float16 s_k[64 * 64];     // [key][d]  (filled by TDM)
  __shared__ _Float16 s_vt[64 * 80];    // [d][key]  (transposed, pad->80)

  int tid  = threadIdx.x;
  int lane = tid & 31;
  int wave = tid >> 5;
  int half = lane >> 4;
  int l16  = lane & 15;
  int kbaseA = half * 16;
  int kbaseB = half * 16;

  const int nblk = NQ / 16;
  int n0 = (blockIdx.x % nblk) * 16;
  int h  = (blockIdx.x / nblk) % NHEAD;
  int b  = blockIdx.x / (nblk * NHEAD);

  const _Float16* Qb = Qf + ((size_t)(b * NQ + n0)) * DMODEL + h * DHEAD;
  const _Float16* Kb = Kf + ((size_t)b * MK) * DMODEL + h * DHEAD;
  const _Float16* Vb = Vf + ((size_t)b * MK) * DMODEL + h * DHEAD;

  for (int i = tid; i < 16 * NR; i += 128) {
    int rr = i / NR, kk = i % NR;
    s_qer[rr][kk] = QErg[(((size_t)(b * NHEAD + h)) * NQ + n0 + rr) * NR + kk];
  }
  __syncthreads();

  // Q A-frags (Dh=64 -> two 32-wide k-steps), kept in registers for all chunks
  v16h qa0 = load_afrag(Qb + (size_t)l16 * DMODEL + kbaseA);
  v16h qa1 = load_afrag(Qb + (size_t)l16 * DMODEL + 32 + kbaseA);

  v8f o = {0.f, 0.f, 0.f, 0.f, 0.f, 0.f, 0.f, 0.f};
  float run_max = -3.0e38f, run_sum = 0.f;

  int dcol = wave * 16 + l16;               // lane's column within the 64 tile
  int nchunk = (8 * (n0 + 15)) / 64 + 1;    // skip fully masked key chunks
  unsigned sk_lds = (unsigned)(uintptr_t)(void*)&s_k[0];  // LDS byte offset

  for (int ch = 0; ch < nchunk; ++ch) {
    int m0 = ch * 64;

    // ---- stage K chunk [64 keys x 64 d] into s_k via Tensor Data Mover ----
#if __has_builtin(__builtin_amdgcn_tensor_load_to_lds)
    if (wave == 0) {
      unsigned long long ga =
          (unsigned long long)(uintptr_t)(Kb + (size_t)m0 * DMODEL);
      v4u g0;
      g0[0] = 1u;                                   // count=1, user mode
      g0[1] = sk_lds;                               // lds_addr
      g0[2] = (unsigned)(ga & 0xFFFFFFFFu);         // global_addr[31:0]
      g0[3] = (unsigned)((ga >> 32) & 0x01FFFFFFu)  // global_addr[56:32]
              | (2u << 30);                         // type=2 (image)
      v8i g1;
      g1[0] = (1 << 16);       // data_size=1 -> 2 bytes; no multicast
      g1[1] = (64 << 16);      // tensor_dim0[15:0]=64 (d)
      g1[2] = (64 << 16);      // tensor_dim0[31:16]=0, tensor_dim1[15:0]=64
      g1[3] = (64 << 16);      // tensor_dim1[31:16]=0, tile_dim0=64
      g1[4] = 64;              // tile_dim1=64, tile_dim2=0
      g1[5] = DMODEL;          // tensor_dim0_stride low = 512
      g1[6] = 0;
      g1[7] = 0;
      v4i z4 = {0, 0, 0, 0};
#if __clang_major__ >= 23
      v8i z8 = {0, 0, 0, 0, 0, 0, 0, 0};
      __builtin_amdgcn_tensor_load_to_lds(g0, g1, z4, z4, z8, 0);
#else
      __builtin_amdgcn_tensor_load_to_lds(g0, g1, z4, z4, 0);
#endif
    }
#else
    // fallback: cooperative linear copy
    for (int i = tid * 8; i < 64 * 64; i += 128 * 8) {
      int key = i >> 6, d = i & 63;
      *(v8h*)&s_k[i] = *(const v8h*)(Kb + (size_t)(m0 + key) * DMODEL + d);
    }
#endif

    // ---- stage V chunk transposed: s_vt[d][key] (coalesced 16B loads) ----
    {
      int mrow = tid >> 3;           // 0..15
      int dsub = (tid & 7) * 8;      // 0..56
#pragma unroll
      for (int pass = 0; pass < 4; ++pass) {
        int m = mrow + pass * 16;
        v8h vv = *(const v8h*)(Vb + (size_t)(m0 + m) * DMODEL + dsub);
#pragma unroll
        for (int j = 0; j < 8; ++j) s_vt[(dsub + j) * 80 + m] = vv[j];
      }
    }
    if (ch + 1 < nchunk)
      __builtin_prefetch(Vb + (size_t)(m0 + 64) * DMODEL, 0, 1);

#if __has_builtin(__builtin_amdgcn_tensor_load_to_lds)
    if (wave == 0) __builtin_amdgcn_s_wait_tensorcnt(0);
#endif
    __syncthreads();   // K (TDM) + V (waves) staged

    // ---- S = Q K^T for this wave's 16 key columns (K from LDS) ----
    v16h kb0 = *(const v16h*)(&s_k[dcol * 64 + kbaseB]);
    v16h kb1 = *(const v16h*)(&s_k[dcol * 64 + 32 + kbaseB]);
    v8f c = {0.f, 0.f, 0.f, 0.f, 0.f, 0.f, 0.f, 0.f};
    c = wmma16(qa0, kb0, c);
    c = wmma16(qa1, kb1, c);

    // ---- relative bias: 31-tap Gaussian softmax via incremental products ---
    int mcol = m0 + dcol;
    float jm = (float)mcol * (1.0f / MK);
#pragma unroll
    for (int v = 0; v < 8; ++v) {
      int rloc = v + half * 8;
      int n = n0 + rloc;
      float alpha = 15.5f * ((float)n * (1.0f / NQ) - jm + 1.0f); // in (0,31)
      int r0 = (int)floorf(alpha + 0.5f);
      r0 = min(30, max(0, r0));
      float d0 = (float)r0 - alpha;
      float w0 = __expf(-d0 * d0 * SIGMA_INV);
      float num = 0.f, den = 0.f;
      // upward: w(r+1) = w(r)*u, u *= exp(-2/sigma)
      float wu = w0;
      float uu = __expf(-(2.f * d0 + 1.f) * SIGMA_INV);
      for (int r = r0; r < NR; ++r) {
        num += s_qer[rloc][r] * wu;
        den += wu;
        wu *= uu;
        uu *= EXP_M2SIG;
      }
      // downward: w(r-1) = w(r)*u, u *= exp(-2/sigma)
      float ud = __expf((2.f * d0 - 1.f) * SIGMA_INV);
      float wd = w0 * ud;
      ud *= EXP_M2SIG;
      for (int r = r0 - 1; r >= 0; --r) {
        num += s_qer[rloc][r] * wd;
        den += wd;
        wd *= ud;
        ud *= EXP_M2SIG;
      }
      float s = (c[v] + num / den) * 0.125f;   // / sqrt(64)
      if (8 * n < mcol) s = NEGINF;            // mask: keep if n/N >= m/M
      s_s[rloc][dcol] = s;
    }
    __syncthreads();

    // ---- online softmax: one thread per query row ----
    if (tid < 16) {
      float cmax = NEGINF;
      for (int cc = 0; cc < 64; ++cc) cmax = fmaxf(cmax, s_s[tid][cc]);
      float nm = fmaxf(run_max, cmax);
      float corr = __expf(run_max - nm);
      float ssum = 0.f;
      for (int cc = 0; cc < 64; ++cc) {
        float p = __expf(s_s[tid][cc] - nm);
        s_p[tid * 64 + cc] = (_Float16)p;
        ssum += p;
      }
      run_sum = run_sum * corr + ssum;
      run_max = nm;
      s_corr[tid] = corr;
    }
    __syncthreads();

    // ---- rescale O, then O += P @ V (V from transposed LDS tile) ----
#pragma unroll
    for (int v = 0; v < 8; ++v) o[v] *= s_corr[v + half * 8];

    v16h pa0 = load_afrag(s_p + l16 * 64 + kbaseA);
    v16h pa1 = load_afrag(s_p + l16 * 64 + 32 + kbaseA);
    v16h vb0 = *(const v16h*)(&s_vt[dcol * 80 + kbaseB]);
    v16h vb1 = *(const v16h*)(&s_vt[dcol * 80 + 32 + kbaseB]);
    o = wmma16(pa0, vb0, o);
    o = wmma16(pa1, vb1, o);
    __syncthreads();   // protect staging buffers before next chunk
  }

  if (tid < 16) s_corr[tid] = 1.0f / run_sum;
  __syncthreads();
#pragma unroll
  for (int v = 0; v < 8; ++v) {
    int rloc = v + half * 8;
    float val = o[v] * s_corr[rloc];
    Og[((size_t)(b * NQ + n0 + rloc)) * DMODEL + h * DHEAD + dcol] =
        (_Float16)val;
  }
}

// ---------------------------------------------------------------------------
extern "C" void kernel_launch(void* const* d_in, const int* in_sizes, int n_in,
                              void* d_out, int out_size, void* d_ws,
                              size_t ws_size, hipStream_t stream) {
  const float* x_latent = (const float*)d_in[0];
  const float* x_input  = (const float*)d_in[1];
  const float* Wk = (const float*)d_in[2];
  const float* bk = (const float*)d_in[3];
  const float* Wv = (const float*)d_in[4];
  const float* bv = (const float*)d_in[5];
  const float* Wq = (const float*)d_in[6];
  const float* bq = (const float*)d_in[7];
  const float* Wo = (const float*)d_in[8];
  const float* bo = (const float*)d_in[9];
  const float* Er = (const float*)d_in[10];
  float* out = (float*)d_out;

  char* ws = (char*)d_ws;
  size_t off = 0;
  auto alloc = [&](size_t bytes) -> char* {
    char* p = ws + off;
    off += bytes;
    off = (off + 255) & ~(size_t)255;
    return p;
  };

  _Float16* Xin  = (_Float16*)alloc((size_t)BSZ * MK * DMODEL * 2);
  _Float16* Xlat = (_Float16*)alloc((size_t)BSZ * NQ * DMODEL * 2);
  _Float16* Wkt  = (_Float16*)alloc((size_t)DMODEL * DMODEL * 2);
  _Float16* Wvt  = (_Float16*)alloc((size_t)DMODEL * DMODEL * 2);
  _Float16* Wqt  = (_Float16*)alloc((size_t)DMODEL * DMODEL * 2);
  _Float16* Wot  = (_Float16*)alloc((size_t)DMODEL * DMODEL * 2);
  _Float16* Kf   = (_Float16*)alloc((size_t)BSZ * MK * DMODEL * 2);
  _Float16* Vf   = (_Float16*)alloc((size_t)BSZ * MK * DMODEL * 2);
  _Float16* Qf   = (_Float16*)alloc((size_t)BSZ * NQ * DMODEL * 2);
  float*    QErg = (float*)alloc((size_t)BSZ * NHEAD * NQ * NR * 4);
  _Float16* Attn = (_Float16*)alloc((size_t)BSZ * NQ * DMODEL * 2);
  (void)ws_size; (void)in_sizes; (void)n_in; (void)out_size;

  // 1) convert activations to f16
  {
    int n = BSZ * MK * DMODEL;
    cvt_f16_kernel<<<(n / 4 + 255) / 256, 256, 0, stream>>>(x_input, Xin, n);
  }
  {
    int n = BSZ * NQ * DMODEL;
    cvt_f16_kernel<<<(n / 4 + 255) / 256, 256, 0, stream>>>(x_latent, Xlat, n);
  }

  // 2) transpose + convert weights
  dim3 tg(DMODEL / 32, DMODEL / 32);
  wtrans_kernel<<<tg, 256, 0, stream>>>(Wk, Wkt);
  wtrans_kernel<<<tg, 256, 0, stream>>>(Wv, Wvt);
  wtrans_kernel<<<tg, 256, 0, stream>>>(Wq, Wqt);
  wtrans_kernel<<<tg, 256, 0, stream>>>(Wo, Wot);

  // 3) projections (WMMA)
  gemm_proj<false><<<dim3(BSZ * MK / 16, DMODEL / 64), 128, 0, stream>>>(
      Xin, Wkt, bk, Kf, DMODEL, DMODEL);
  gemm_proj<false><<<dim3(BSZ * MK / 16, DMODEL / 64), 128, 0, stream>>>(
      Xin, Wvt, bv, Vf, DMODEL, DMODEL);
  gemm_proj<false><<<dim3(BSZ * NQ / 16, DMODEL / 64), 128, 0, stream>>>(
      Xlat, Wqt, bq, Qf, DMODEL, DMODEL);

  // 4) QEr
  qer_kernel<<<BSZ * NHEAD * NQ, 32, 0, stream>>>(Qf, Er, QErg);

  // 5) fused attention (WMMA flash + TDM K staging)
  attn_kernel<<<BSZ * NHEAD * (NQ / 16), 128, 0, stream>>>(Qf, Kf, Vf, QErg,
                                                           Attn);

  // 6) output projection (WMMA, f32 out)
  gemm_proj<true><<<dim3(BSZ * NQ / 16, DMODEL / 64), 128, 0, stream>>>(
      Attn, Wot, bo, out, DMODEL, DMODEL);
}
